// CoreModel_43379169689847
// MI455X (gfx1250) — matrix operand order repaired and checked by reading, not compiled, not run on hardware
//
#include <hip/hip_runtime.h>
#include <hip/hip_bf16.h>

#define B_ 2048
#define TM 128
#define TN 128
#define TK 32

typedef __attribute__((ext_vector_type(16))) __bf16 v16bf;
typedef __attribute__((ext_vector_type(8)))  __bf16 v8bf;
typedef __attribute__((ext_vector_type(8)))  float  v8f;

// ---------------------------------------------------------------------------
// Generic bf16-WMMA GEMM:  C = epilogue( alpha * (A @ B) )
//   A: logical [M, Ktot] fp32, up to 3 row-major K-segments (concat inputs).
//   B: fp32, [Ktot, N] row-major (transB=0) or [N, Ktot] row-major (transB=1).
//   Epilogue: *alpha, +bias[n], act (0 none, 1 tanh, 2 sigmoid), +resid[m,n].
// Tile: 128x128 per WG (8 waves, 2x4), each wave 64x32 via 8 16x16x32 WMMAs.
// Double-buffered LDS pipeline: global loads for tile k+1 are issued before
// the WMMA burst for tile k; one workgroup barrier per K-step.
// Requires M%128==0, N%128==0, all k_s%32==0 (true for every call here).
// ---------------------------------------------------------------------------
__global__ __launch_bounds__(256) void gemm_wmma_bf16(
    const float* __restrict__ A0, const float* __restrict__ A1,
    const float* __restrict__ A2, int k0, int k1, int k2,
    const float* __restrict__ Bm, int ldb, int transB,
    const float* __restrict__ bias, const float* __restrict__ resid,
    float* __restrict__ C, int N, float alpha, int act)
{
  __shared__ __align__(64) __bf16 As[2][TM][TK];   // [buf][m][k]
  __shared__ __align__(64) __bf16 Bs[2][TN][TK];   // [buf][n][k]  (N-major!)

  const int t       = threadIdx.x;
  const int lane    = t & 31;
  const int wave    = t >> 5;
  const int wm      = wave >> 2;        // 0..1
  const int wn      = wave & 3;         // 0..3
  const int halfsel = lane >> 4;        // 0|1
  const int lanelo  = lane & 15;
  const int tileM   = blockIdx.y * TM;
  const int tileN   = blockIdx.x * TN;
  const int Ktot    = k0 + k1 + k2;

  v8f acc[4][2];
  #pragma unroll
  for (int m = 0; m < 4; ++m)
    #pragma unroll
    for (int n = 0; n < 2; ++n)
      #pragma unroll
      for (int i = 0; i < 8; ++i) acc[m][n][i] = 0.0f;

  // staging maps: each thread moves 16 contiguous fp32 for A and for B
  const int arow = t >> 1;            // A: [m][k] tile row
  const int acol = (t & 1) * 16;
  const int bn   = t >> 1;            // B transposed path: row of [n][k]
  const int bc   = (t & 1) * 16;
  const int bk   = t >> 3;            // B normal path: k-line, 16 n's
  const int bn0  = (t & 7) * 16;

  // global loads for the K-tile starting at kt -> registers
  auto ld_tile = [&](int kt, float* aR, float* bR) {
    const float* Ap; int kk, lda;
    if (kt < k0)           { Ap = A0; kk = kt;           lda = k0; }
    else if (kt < k0 + k1) { Ap = A1; kk = kt - k0;      lda = k1; }
    else                   { Ap = A2; kk = kt - k0 - k1; lda = k2; }
    const float* asrc = Ap + (size_t)(tileM + arow) * lda + kk + acol;
    #pragma unroll
    for (int i = 0; i < 16; ++i) aR[i] = asrc[i];
    const float* bsrc = transB
        ? Bm + (size_t)(tileN + bn) * ldb + kt + bc
        : Bm + (size_t)(kt + bk) * ldb + tileN + bn0;
    #pragma unroll
    for (int i = 0; i < 16; ++i) bR[i] = bsrc[i];
  };

  // registers -> LDS buffer `buf`, with fp32 -> bf16 conversion
  auto st_tile = [&](int buf, const float* aR, const float* bR) {
    v8bf a0, a1;
    #pragma unroll
    for (int i = 0; i < 8; ++i) { a0[i] = (__bf16)aR[i]; a1[i] = (__bf16)aR[8 + i]; }
    *(v8bf*)&As[buf][arow][acol]     = a0;
    *(v8bf*)&As[buf][arow][acol + 8] = a1;
    if (transB) {
      v8bf b0, b1;
      #pragma unroll
      for (int i = 0; i < 8; ++i) { b0[i] = (__bf16)bR[i]; b1[i] = (__bf16)bR[8 + i]; }
      *(v8bf*)&Bs[buf][bn][bc]     = b0;
      *(v8bf*)&Bs[buf][bn][bc + 8] = b1;
    } else {
      #pragma unroll
      for (int i = 0; i < 16; ++i) Bs[buf][bn0 + i][bk] = (__bf16)bR[i];
    }
  };

  float aR[16], bR[16];
  ld_tile(0, aR, bR);
  st_tile(0, aR, bR);
  __syncthreads();

  int cur = 0;
  for (int kt = 0; kt < Ktot; kt += TK) {
    const bool hasNext = (kt + TK) < Ktot;
    if (hasNext) ld_tile(kt + TK, aR, bR);   // overlap HBM with WMMAs below

    // ---- fragments per ISA 16-bit WMMA layouts ----
    // A 16x32: lanes 0-15 hold K [0..7]+[16..23]; lanes 16-31 K [8..15]+[24..31]
    v16bf af[4];
    const int ka0 = halfsel * 8;
    #pragma unroll
    for (int m = 0; m < 4; ++m) {
      const int row = wm * 64 + m * 16 + lanelo;
      v8bf lo = *(const v8bf*)&As[cur][row][ka0];
      v8bf hi = *(const v8bf*)&As[cur][row][ka0 + 16];
      #pragma unroll
      for (int i = 0; i < 8; ++i) { af[m][i] = lo[i]; af[m][i + 8] = hi[i]; }
    }
    // B 32x16: lane = half*16 + n; lanes 0-15 K [0..7]+[16..23]; lanes 16-31 K [8..15]+[24..31]
    v16bf bfr[2];
    const int kb0 = halfsel * 8;
    #pragma unroll
    for (int n = 0; n < 2; ++n) {
      const int col = wn * 32 + n * 16 + lanelo;
      v8bf lo = *(const v8bf*)&Bs[cur][col][kb0];
      v8bf hi = *(const v8bf*)&Bs[cur][col][kb0 + 16];
      #pragma unroll
      for (int i = 0; i < 8; ++i) { bfr[n][i] = lo[i]; bfr[n][i + 8] = hi[i]; }
    }

    #pragma unroll
    for (int m = 0; m < 4; ++m)
      #pragma unroll
      for (int n = 0; n < 2; ++n)
        acc[m][n] = __builtin_amdgcn_wmma_f32_16x16x32_bf16(
            false, af[m], false, bfr[n], (short)0, acc[m][n], false, false);

    if (hasNext) {
      st_tile(cur ^ 1, aR, bR);
      __syncthreads();
      cur ^= 1;
    }
  }

  // ---- epilogue (C/D layout: VGPR r -> M = r + 8*halfsel, N = lanelo) ----
  #pragma unroll
  for (int m = 0; m < 4; ++m)
    #pragma unroll
    for (int n = 0; n < 2; ++n) {
      const int gn = tileN + wn * 32 + n * 16 + lanelo;
      const float bv = bias ? bias[gn] : 0.0f;
      #pragma unroll
      for (int r = 0; r < 8; ++r) {
        const int gm = tileM + wm * 64 + m * 16 + r + 8 * halfsel;
        float v = acc[m][n][r] * alpha + bv;
        if (act == 1)      v = tanhf(v);
        else if (act == 2) v = 1.0f / (1.0f + __expf(-v));
        if (resid) v += resid[(size_t)gm * N + gn];
        C[(size_t)gm * N + gn] = v;
      }
    }
}

// ---------------------------------------------------------------------------
// Row LayerNorm (population var) + ReLU, in place. One block per row.
// ---------------------------------------------------------------------------
__global__ __launch_bounds__(256) void ln_relu_kernel(
    float* __restrict__ x, const float* __restrict__ scale,
    const float* __restrict__ bias, int D)
{
  __shared__ float red[256];
  float* row = x + (size_t)blockIdx.x * D;
  const int t = threadIdx.x;
  float s = 0.f, ss = 0.f;
  for (int i = t; i < D; i += 256) { float v = row[i]; s += v; ss += v * v; }
  red[t] = s; __syncthreads();
  for (int o = 128; o > 0; o >>= 1) { if (t < o) red[t] += red[t + o]; __syncthreads(); }
  const float mu = red[0] / D; __syncthreads();
  red[t] = ss; __syncthreads();
  for (int o = 128; o > 0; o >>= 1) { if (t < o) red[t] += red[t + o]; __syncthreads(); }
  const float var = red[0] / D - mu * mu;
  const float inv = rsqrtf(var + 1e-6f);
  for (int i = t; i < D; i += 256) {
    float v = (row[i] - mu) * inv * scale[i] + bias[i];
    row[i] = v > 0.f ? v : 0.f;
  }
}

// ---------------------------------------------------------------------------
// Row softmax, in place. One block per row.
// ---------------------------------------------------------------------------
__global__ __launch_bounds__(256) void softmax_kernel(float* __restrict__ x, int N)
{
  __shared__ float red[256];
  float* row = x + (size_t)blockIdx.x * N;
  const int t = threadIdx.x;
  float mx = -3.0e38f;
  for (int i = t; i < N; i += 256) mx = fmaxf(mx, row[i]);
  red[t] = mx; __syncthreads();
  for (int o = 128; o > 0; o >>= 1) { if (t < o) red[t] = fmaxf(red[t], red[t + o]); __syncthreads(); }
  const float m = red[0]; __syncthreads();
  float s = 0.f;
  for (int i = t; i < N; i += 256) { float e = __expf(row[i] - m); row[i] = e; s += e; }
  red[t] = s; __syncthreads();
  for (int o = 128; o > 0; o >>= 1) { if (t < o) red[t] += red[t + o]; __syncthreads(); }
  const float inv = 1.0f / red[0];
  for (int i = t; i < N; i += 256) row[i] *= inv;
}

// ---------------------------------------------------------------------------
// mix = softmax( concat(ctx0,ctx1,ctx2) @ Wmix + bmix )   (N_LEV = 3)
// One block (128 thr) per batch row.
// ---------------------------------------------------------------------------
__global__ __launch_bounds__(128) void mix_kernel(
    const float* __restrict__ c0, const float* __restrict__ c1,
    const float* __restrict__ c2, const float* __restrict__ Wmix,
    const float* __restrict__ bmix, float* __restrict__ mix)
{
  __shared__ float red[3][128];
  const int b = blockIdx.x, t = threadIdx.x;
  const float* rows[3] = { c0 + (size_t)b * 1024, c1 + (size_t)b * 1024, c2 + (size_t)b * 1024 };
  float a0 = 0.f, a1 = 0.f, a2 = 0.f;
  for (int seg = 0; seg < 3; ++seg) {
    const float* r = rows[seg];
    const float* w = Wmix + (size_t)seg * 1024 * 3;
    for (int i = t; i < 1024; i += 128) {
      const float v = r[i];
      a0 += v * w[i * 3 + 0]; a1 += v * w[i * 3 + 1]; a2 += v * w[i * 3 + 2];
    }
  }
  red[0][t] = a0; red[1][t] = a1; red[2][t] = a2; __syncthreads();
  for (int o = 64; o > 0; o >>= 1) {
    if (t < o) { red[0][t] += red[0][t + o]; red[1][t] += red[1][t + o]; red[2][t] += red[2][t + o]; }
    __syncthreads();
  }
  if (t == 0) {
    const float l0 = red[0][0] + bmix[0], l1 = red[1][0] + bmix[1], l2 = red[2][0] + bmix[2];
    const float m  = fmaxf(l0, fmaxf(l1, l2));
    const float e0 = __expf(l0 - m), e1 = __expf(l1 - m), e2 = __expf(l2 - m);
    const float inv = 1.0f / (e0 + e1 + e2);
    mix[b * 3 + 0] = e0 * inv; mix[b * 3 + 1] = e1 * inv; mix[b * 3 + 2] = e2 * inv;
  }
}

// c_t = sum_l mix[:,l] * ctx_l
__global__ __launch_bounds__(256) void combine_kernel(
    const float* __restrict__ mix, const float* __restrict__ c0,
    const float* __restrict__ c1, const float* __restrict__ c2,
    float* __restrict__ c)
{
  const size_t idx = (size_t)blockIdx.x * 256 + threadIdx.x;
  const int b = (int)(idx >> 10);  // /1024
  c[idx] = mix[b * 3 + 0] * c0[idx] + mix[b * 3 + 1] * c1[idx] + mix[b * 3 + 2] * c2[idx];
}

// w_t = mod * (tanh(A_diag)*0.9 * w_prev + wlin2) + wB      (mod pre-tanh'ed)
__global__ __launch_bounds__(256) void wave_kernel(
    const float* __restrict__ mod, const float* __restrict__ A_diag,
    const float* __restrict__ w_prev, const float* __restrict__ wlin2,
    const float* __restrict__ wB, float* __restrict__ w_out)
{
  const size_t idx = (size_t)blockIdx.x * 256 + threadIdx.x;
  const int j = (int)(idx & 511);
  const float Ad = tanhf(A_diag[j]) * 0.9f;
  w_out[idx] = mod[idx] * (Ad * w_prev[idx] + wlin2[idx]) + wB[idx];
}

// s_t = clip(s_prev + g*Pp + (1-g)*Wp, +-10)
__global__ __launch_bounds__(256) void final_kernel(
    const float* __restrict__ s_prev, const float* __restrict__ g,
    const float* __restrict__ Pp, const float* __restrict__ Wp,
    float* __restrict__ s_out)
{
  const size_t idx = (size_t)blockIdx.x * 256 + threadIdx.x;
  const float gg = g[idx];
  float v = s_prev[idx] + gg * Pp[idx] + (1.0f - gg) * Wp[idx];
  v = fminf(10.0f, fmaxf(-10.0f, v));
  s_out[idx] = v;
}

// ---------------------------------------------------------------------------
extern "C" void kernel_launch(void* const* d_in, const int* in_sizes, int n_in,
                              void* d_out, int out_size, void* d_ws, size_t ws_size,
                              hipStream_t stream) {
  const float* s_prev = (const float*)d_in[0];
  const float* w_prev = (const float*)d_in[1];
  const float* p_prev = (const float*)d_in[2];
  const float* e_t    = (const float*)d_in[3];
  const float* Mm[3]  = {(const float*)d_in[4], (const float*)d_in[5], (const float*)d_in[6]};
  const float* Kk[3]  = {(const float*)d_in[7], (const float*)d_in[8], (const float*)d_in[9]};
  const float* Wq1 = (const float*)d_in[10]; const float* bq1  = (const float*)d_in[11];
  const float* lnqs= (const float*)d_in[12]; const float* lnqb = (const float*)d_in[13];
  const float* Wq2 = (const float*)d_in[14]; const float* bq2  = (const float*)d_in[15];
  const float* Wl[3]= {(const float*)d_in[16], (const float*)d_in[17], (const float*)d_in[18]};
  const float* Wmx = (const float*)d_in[19]; const float* bmx  = (const float*)d_in[20];
  const float* Wz  = (const float*)d_in[21]; const float* bz   = (const float*)d_in[22];
  const float* A_d = (const float*)d_in[23];
  const float* A_U = (const float*)d_in[24]; const float* A_V  = (const float*)d_in[25];
  const float* Wam = (const float*)d_in[26]; const float* bam  = (const float*)d_in[27];
  const float* WBm = (const float*)d_in[28]; const float* bBv  = (const float*)d_in[29];
  const float* Wp1 = (const float*)d_in[30]; const float* bp1  = (const float*)d_in[31];
  const float* lnps= (const float*)d_in[32]; const float* lnpb = (const float*)d_in[33];
  const float* Wp2 = (const float*)d_in[34]; const float* bp2  = (const float*)d_in[35];
  const float* Wg  = (const float*)d_in[36]; const float* bg   = (const float*)d_in[37];
  const float* U_p = (const float*)d_in[38]; const float* U_w  = (const float*)d_in[39];

  float* ws = (float*)d_ws;
  size_t off = 0;
  auto alloc = [&](size_t n) { float* p = ws + off; off += n; return p; };
  float* t2048 = alloc((size_t)B_ * 2048);   // q1raw/qh, later p1raw/ph (in-place LN)
  float* qbuf  = alloc((size_t)B_ * 256);
  float* attn  = alloc((size_t)B_ * 4096);   // reused per level
  float* ctxm  = alloc((size_t)B_ * 1024);   // reused per level
  float* ctx0  = alloc((size_t)B_ * 1024);
  float* ctx1  = alloc((size_t)B_ * 1024);
  float* ctx2  = alloc((size_t)B_ * 1024);
  float* mixb  = alloc((size_t)B_ * 4);
  float* cbuf  = alloc((size_t)B_ * 1024);
  float* zbuf  = alloc((size_t)B_ * 1024);
  float* wtmp  = alloc((size_t)B_ * 128);
  float* wlin2 = alloc((size_t)B_ * 512);
  float* amod  = alloc((size_t)B_ * 512);
  float* wBb   = alloc((size_t)B_ * 512);
  float* gbuf  = alloc((size_t)B_ * 1024);
  float* Ppb   = alloc((size_t)B_ * 1024);
  float* Wpb   = alloc((size_t)B_ * 1024);
  (void)ws_size; (void)in_sizes; (void)n_in; (void)out_size;

  float* s_out = (float*)d_out;                    // [B, 1024]
  float* w_out = s_out + (size_t)B_ * 1024;        // [B, 512]
  float* p_out = w_out + (size_t)B_ * 512;         // [B, 1024]

  const dim3 blk(256);
  float* ctxs[3] = { ctx0, ctx1, ctx2 };
  auto gemm = [&](const float* a0, const float* a1, const float* a2,
                  int k0, int k1, int k2, const float* Bp, int ldb, int tb,
                  const float* bias, const float* resid, float* Cp, int N,
                  float alpha, int act) {
    dim3 grid(N / TN, B_ / TM);
    gemm_wmma_bf16<<<grid, blk, 0, stream>>>(a0, a1, a2, k0, k1, k2, Bp, ldb, tb,
                                             bias, resid, Cp, N, alpha, act);
  };

  // ---- CMS read path ----
  gemm(s_prev, e_t, nullptr, 1024, 1024, 0, Wq1, 2048, 0, bq1, nullptr, t2048, 2048, 1.f, 0);
  ln_relu_kernel<<<B_, 256, 0, stream>>>(t2048, lnqs, lnqb, 2048);
  gemm(t2048, nullptr, nullptr, 2048, 0, 0, Wq2, 256, 0, bq2, nullptr, qbuf, 256, 1.f, 0);
  for (int l = 0; l < 3; ++l) {
    gemm(qbuf, nullptr, nullptr, 256, 0, 0, Kk[l], 256, 1, nullptr, nullptr, attn, 4096, 0.0625f, 0);
    softmax_kernel<<<B_, 256, 0, stream>>>(attn, 4096);
    gemm(attn, nullptr, nullptr, 4096, 0, 0, Mm[l], 1024, 0, nullptr, nullptr, ctxm, 1024, 1.f, 0);
    gemm(ctxm, nullptr, nullptr, 1024, 0, 0, Wl[l], 1024, 0, nullptr, nullptr, ctxs[l], 1024, 1.f, 0);
  }
  mix_kernel<<<B_, 128, 0, stream>>>(ctx0, ctx1, ctx2, Wmx, bmx, mixb);
  combine_kernel<<<(B_ * 1024) / 256, blk, 0, stream>>>(mixb, ctx0, ctx1, ctx2, cbuf);

  // ---- HOPE core: wave ----
  gemm(e_t, nullptr, nullptr, 1024, 0, 0, Wz, 1024, 0, bz, nullptr, zbuf, 1024, 1.f, 0);
  gemm(w_prev, nullptr, nullptr, 512, 0, 0, A_V, 128, 0, nullptr, nullptr, wtmp, 128, 1.f, 0);
  gemm(wtmp, nullptr, nullptr, 128, 0, 0, A_U, 128, 1, nullptr, nullptr, wlin2, 512, 1.f, 0);
  gemm(cbuf, nullptr, nullptr, 1024, 0, 0, Wam, 512, 0, bam, nullptr, amod, 512, 1.f, 1);   // tanh
  gemm(cbuf, zbuf, nullptr, 1024, 1024, 0, WBm, 512, 0, bBv, nullptr, wBb, 512, 1.f, 0);
  wave_kernel<<<(B_ * 512) / 256, blk, 0, stream>>>(amod, A_d, w_prev, wlin2, wBb, w_out);

  // ---- particle ----
  gemm(p_prev, cbuf, nullptr, 1024, 1024, 0, Wp1, 2048, 0, bp1, nullptr, t2048, 2048, 1.f, 0);
  ln_relu_kernel<<<B_, 256, 0, stream>>>(t2048, lnps, lnpb, 2048);
  gemm(t2048, nullptr, nullptr, 2048, 0, 0, Wp2, 1024, 0, bp2, p_prev, p_out, 1024, 1.f, 0);

  // ---- gated fast-state mix ----
  gemm(s_prev, e_t, cbuf, 1024, 1024, 1024, Wg, 1024, 0, bg, nullptr, gbuf, 1024, 1.f, 2);  // sigmoid
  gemm(p_out, nullptr, nullptr, 1024, 0, 0, U_p, 1024, 1, nullptr, nullptr, Ppb, 1024, 1.f, 0);
  gemm(w_out, nullptr, nullptr, 512, 0, 0, U_w, 512, 1, nullptr, nullptr, Wpb, 1024, 1.f, 0);
  final_kernel<<<(B_ * 1024) / 256, blk, 0, stream>>>(s_prev, gbuf, Ppb, Wpb, s_out);
}
